// MultiHeadAttention_18262200942835
// MI455X (gfx1250) — matrix operand order repaired and checked by reading, not compiled, not run on hardware
//
#include <hip/hip_runtime.h>
#include <hip/hip_bf16.h>
#include <stdint.h>

// ---------------------------------------------------------------------------
// MI455X (gfx1250) multi-head attention, bf16 WMMA pipeline.
//   B=4, T=2048, S=1024, H=16, D=64.
// Pipeline: cvt(f32->bf16) -> fused QKV GEMM (V stored transposed)
//           -> causal flash attention (paired k-tiles, K=32 PV WMMAs)
//           -> output projection GEMM.
// ---------------------------------------------------------------------------

#define N_CTX   2048
#define N_STATE 1024
#define N_HEAD  16
#define HDIM    64
#define BATCH   4
#define BT      (BATCH * N_CTX)                            // 8192 rows
#define QEL     ((size_t)BATCH * N_HEAD * N_CTX * HDIM)    // 8,388,608 elems
#define WEL     ((size_t)N_STATE * N_STATE)                // 1,048,576 elems
#define QK_SCALE 0.35355339059327379f                      // 64^-0.25

typedef __bf16 v16bf __attribute__((ext_vector_type(16)));
typedef __bf16 v4bf  __attribute__((ext_vector_type(4)));
typedef float  v8f   __attribute__((ext_vector_type(8)));
typedef float  v4f   __attribute__((ext_vector_type(4)));

// ---------------------------------------------------------------------------
// f32 -> bf16 bulk conversion (native packed cvt via convertvector)
// ---------------------------------------------------------------------------
__global__ __launch_bounds__(256) void cvt_kernel(const float* __restrict__ src,
                                                  __bf16* __restrict__ dst,
                                                  int n4) {
  const int i = blockIdx.x * blockDim.x + threadIdx.x;
  if (i < n4) {
    v4f v = ((const v4f*)src)[i];
    ((v4bf*)dst)[i] = __builtin_convertvector(v, v4bf);
  }
}

// ---- fragment loaders (CDNA5 16-bit WMMA VGPR layouts, wave32) ------------
// A 16x32 (MxK), row-major bf16 source.
// lane<16: row=lane, j0..7 -> K=k0+j, j8..15 -> K=k0+16+j
// lane>=16: row=lane-16, K ranges shifted by +8.
static __device__ __forceinline__ v16bf load_a_bf16(const __bf16* __restrict__ A,
                                                    int lda, int m0, int k0, int lane) {
  const int half = lane >> 4;
  const __bf16* p = A + (size_t)(m0 + (lane & 15)) * lda + k0 + half * 8;
  v16bf a;
#pragma unroll
  for (int j = 0; j < 8; ++j) a[j] = p[j];
#pragma unroll
  for (int j = 0; j < 8; ++j) a[8 + j] = p[16 + j];
  return a;
}
// B 32x16 (KxN) where B[k][n] = M[n][k], M row-major [N,K] (e.g. W, or V^T).
// lane: n = lane&15, j0..7 -> K = k0 + (lane>>4)*8 + j, j8..15 -> K = k0 + 16 + (lane>>4)*8 + (j-8)
static __device__ __forceinline__ v16bf load_bt_bf16(const __bf16* __restrict__ W,
                                                     int ldw, int n0, int k0, int lane) {
  const __bf16* p = W + (size_t)(n0 + (lane & 15)) * ldw + k0 + (lane >> 4) * 8;
  v16bf b;
#pragma unroll
  for (int j = 0; j < 16; ++j) b[j] = p[j + (j >> 3) * 8];
  return b;
}

// ---------------------------------------------------------------------------
// Kernel 1: fused QKV projection, register-blocked.
// One wave = 64x32 tile (4 M-tiles x 2 N-tiles) of [BT, 3S].
// grid = (BT/256, 3S/64) = (32, 48), block = 256 (8 waves)
// Q/K scaled by 64^-0.25, stored bf16 [B,H,T,D]; V stored bf16 [B,H,D,T].
// ---------------------------------------------------------------------------
__global__ __launch_bounds__(256) void qkv_kernel(
    const __bf16* __restrict__ xb,  const __bf16* __restrict__ Wqb,
    const __bf16* __restrict__ Wkb, const __bf16* __restrict__ Wvb,
    const float*  __restrict__ bq,  const float*  __restrict__ bv,
    __bf16* __restrict__ qkv) {            // 3 planes of QEL bf16 each
  const int lane = threadIdx.x & 31;
  const int wave = threadIdx.x >> 5;
  const int half = lane >> 4;
  const int col  = lane & 15;

  const int m_base = blockIdx.x * 256 + (wave & 3) * 64;     // 4 M tiles
  const int n_g    = blockIdx.y * 64 + (wave >> 2) * 32;     // 2 N tiles, [0,3S)
  const int sel    = n_g >> 10;                              // 0=Q 1=K 2=V
  const int n0     = n_g & (N_STATE - 1);

  const __bf16* W = (sel == 0) ? Wqb : ((sel == 1) ? Wkb : Wvb);

  v8f c[4][2];
#pragma unroll
  for (int nt = 0; nt < 2; ++nt) {
    float bias = 0.0f;
    if (sel == 0) bias = bq[n0 + nt * 16 + col];
    else if (sel == 2) bias = bv[n0 + nt * 16 + col];
#pragma unroll
    for (int i = 0; i < 4; ++i)
#pragma unroll
      for (int r = 0; r < 8; ++r) c[i][nt][r] = bias;
  }

  for (int k0 = 0; k0 < N_STATE; k0 += 32) {
    __builtin_prefetch(W + (size_t)(n0 + col) * N_STATE + k0 + 128, 0, 0);
    v16bf a[4];
#pragma unroll
    for (int i = 0; i < 4; ++i)
      a[i] = load_a_bf16(xb, N_STATE, m_base + 16 * i, k0, lane);
#pragma unroll
    for (int nt = 0; nt < 2; ++nt) {
      const v16bf b = load_bt_bf16(W, N_STATE, n0 + nt * 16, k0, lane);
#pragma unroll
      for (int i = 0; i < 4; ++i)
        c[i][nt] = __builtin_amdgcn_wmma_f32_16x16x32_bf16(
            false, a[i], false, b, (short)0, c[i][nt], false, false);
    }
  }

  // ---- store phase: single uniform branch, two fully-unrolled paths ----
  if (sel == 2) {
    // V, transposed layout [B,H,D,T]
    __bf16* dstp = qkv + 2 * QEL;
#pragma unroll
    for (int i = 0; i < 4; ++i)
#pragma unroll
      for (int nt = 0; nt < 2; ++nt)
#pragma unroll
        for (int r = 0; r < 8; ++r) {
          const int m  = m_base + 16 * i + r + 8 * half;
          const int bi = m >> 11;
          const int t  = m & (N_CTX - 1);
          const int nc = n0 + nt * 16 + col;
          const int h  = nc >> 6;
          const int d  = nc & (HDIM - 1);
          dstp[(((size_t)bi * N_HEAD + h) * HDIM + d) * N_CTX + t] =
              (__bf16)c[i][nt][r];
        }
  } else {
    // Q or K, [B,H,T,D], scaled by 64^-0.25
    __bf16* dstp = qkv + (size_t)sel * QEL;
#pragma unroll
    for (int i = 0; i < 4; ++i)
#pragma unroll
      for (int nt = 0; nt < 2; ++nt)
#pragma unroll
        for (int r = 0; r < 8; ++r) {
          const int m  = m_base + 16 * i + r + 8 * half;
          const int bi = m >> 11;
          const int t  = m & (N_CTX - 1);
          const int nc = n0 + nt * 16 + col;
          const int h  = nc >> 6;
          const int d  = nc & (HDIM - 1);
          dstp[(((size_t)bi * N_HEAD + h) * N_CTX + t) * HDIM + d] =
              (__bf16)(c[i][nt][r] * QK_SCALE);
        }
  }
}

// ---------------------------------------------------------------------------
// Kernel 2: causal flash attention, paired k-tiles (32 keys / iteration).
// One wave = one (b, h, 16-row q tile).  Per pair: 4 QK WMMAs + 4 PV WMMAs,
// P (16x32) staged through LDS to re-layout C->A; V read from [B,H,D,T]
// (two contiguous 16B runs per lane).  For even qt the trailing tile is fully masked
// analytically (valid memory, exp -> 0), so no edge handling is needed.
// grid = B*H*(T/16)/8 = 1024, block = 256
// ---------------------------------------------------------------------------
__global__ __launch_bounds__(256) void attn_kernel(
    const __bf16* __restrict__ Q, const __bf16* __restrict__ K,
    const __bf16* __restrict__ Vt, __bf16* __restrict__ wv) {
  __shared__ __bf16 ldsP[8][16][32];

  const int lane = threadIdx.x & 31;
  const int wave = threadIdx.x >> 5;
  const int half = lane >> 4;
  const int col  = lane & 15;

  const int id = blockIdx.x * 8 + wave;
  const int qt = id & 127;
  const int h  = (id >> 7) & 15;
  const int b  = id >> 11;
  const int t0 = qt * 16;

  const size_t bh = ((size_t)b * N_HEAD + h) * N_CTX * HDIM;
  const __bf16* Qb  = Q + bh;
  const __bf16* Kb  = K + bh;
  const __bf16* Vtb = Vt + bh;     // [D, T] per (b,h)

  // hoisted Q fragments (K=0..31, 32..63)
  const v16bf aQ0 = load_a_bf16(Qb, HDIM, t0, 0,  lane);
  const v16bf aQ1 = load_a_bf16(Qb, HDIM, t0, 32, lane);

  float mrun[8], lrun[8], alpha[8];
  v8f o[4];
#pragma unroll
  for (int r = 0; r < 8; ++r) { mrun[r] = -3.0e38f; lrun[r] = 0.0f; }
#pragma unroll
  for (int f = 0; f < 4; ++f)
#pragma unroll
    for (int r = 0; r < 8; ++r) o[f][r] = 0.0f;

  for (int kt = 0; kt <= qt; kt += 2) {   // uniform per wave -> EXEC all ones
    // ---- S tiles for kt and kt+1 (16x32 of scores) ----
    v8f s0, s1;
#pragma unroll
    for (int r = 0; r < 8; ++r) { s0[r] = 0.0f; s1[r] = 0.0f; }
    v16bf bK = load_bt_bf16(Kb, HDIM, kt * 16, 0, lane);
    s0 = __builtin_amdgcn_wmma_f32_16x16x32_bf16(false, aQ0, false, bK,
                                                 (short)0, s0, false, false);
    bK = load_bt_bf16(Kb, HDIM, kt * 16, 32, lane);
    s0 = __builtin_amdgcn_wmma_f32_16x16x32_bf16(false, aQ1, false, bK,
                                                 (short)0, s0, false, false);
    bK = load_bt_bf16(Kb, HDIM, (kt + 1) * 16, 0, lane);
    s1 = __builtin_amdgcn_wmma_f32_16x16x32_bf16(false, aQ0, false, bK,
                                                 (short)0, s1, false, false);
    bK = load_bt_bf16(Kb, HDIM, (kt + 1) * 16, 32, lane);
    s1 = __builtin_amdgcn_wmma_f32_16x16x32_bf16(false, aQ1, false, bK,
                                                 (short)0, s1, false, false);

    // ---- causal mask + online softmax over 32 columns ----
    const int kc0 = kt * 16 + col;
    const int kc1 = kc0 + 16;
#pragma unroll
    for (int r = 0; r < 8; ++r) {
      const int qrow = t0 + r + 8 * half;
      float sv0 = s0[r] + ((kc0 > qrow) ? -1.0e9f : 0.0f);
      float sv1 = s1[r] + ((kc1 > qrow) ? -1.0e9f : 0.0f);
      float rm = fmaxf(sv0, sv1);
#pragma unroll
      for (int off = 1; off < 16; off <<= 1)
        rm = fmaxf(rm, __shfl_xor(rm, off, 32));
      const float mnew = fmaxf(mrun[r], rm);
      const float p0 = __expf(sv0 - mnew);
      const float p1 = __expf(sv1 - mnew);
      float rs = p0 + p1;
#pragma unroll
      for (int off = 1; off < 16; off <<= 1)
        rs += __shfl_xor(rs, off, 32);
      alpha[r] = __expf(mrun[r] - mnew);
      lrun[r]  = lrun[r] * alpha[r] + rs;
      mrun[r]  = mnew;
      ldsP[wave][r + 8 * half][col]      = (__bf16)p0;
      ldsP[wave][r + 8 * half][col + 16] = (__bf16)p1;
    }
    asm volatile("s_wait_dscnt 0" ::: "memory");   // order LDS store -> load

    // ---- rescale running O ----
#pragma unroll
    for (int f = 0; f < 4; ++f)
#pragma unroll
      for (int r = 0; r < 8; ++r) o[f][r] *= alpha[r];

    // ---- P (16x32) as full A fragment ----
    v16bf aP;
    {
      const int prow = lane & 15;
      const int bc   = half * 8;
#pragma unroll
      for (int j = 0; j < 8; ++j) aP[j]     = ldsP[wave][prow][bc + j];
#pragma unroll
      for (int j = 0; j < 8; ++j) aP[8 + j] = ldsP[wave][prow][16 + bc + j];
    }

    // ---- O += P * V  (V^T rows contiguous over keys) ----
#pragma unroll
    for (int f = 0; f < 4; ++f) {
      const v16bf bV = load_bt_bf16(Vtb, N_CTX, f * 16, kt * 16, lane);
      o[f] = __builtin_amdgcn_wmma_f32_16x16x32_bf16(false, aP, false, bV,
                                                     (short)0, o[f], false, false);
    }
  }

  // ---- normalize and store wv (bf16, [BT, S] layout) ----
#pragma unroll
  for (int r = 0; r < 8; ++r) {
    const float inv = 1.0f / lrun[r];
    const int m = t0 + r + 8 * half;
    const size_t rowb = ((size_t)b * N_CTX + m) * N_STATE + h * HDIM;
#pragma unroll
    for (int f = 0; f < 4; ++f)
      wv[rowb + f * 16 + col] = (__bf16)(o[f][r] * inv);
  }
}

// ---------------------------------------------------------------------------
// Kernel 3: output projection  out = wv * Wo^T + bo   (f32 out)
// Same 64x32-per-wave blocking.  grid = (BT/256, S/64) = (32, 16)
// ---------------------------------------------------------------------------
__global__ __launch_bounds__(256) void oproj_kernel(
    const __bf16* __restrict__ wv, const __bf16* __restrict__ Wob,
    const float* __restrict__ bo,  float* __restrict__ out) {
  const int lane = threadIdx.x & 31;
  const int wave = threadIdx.x >> 5;
  const int half = lane >> 4;
  const int col  = lane & 15;

  const int m_base = blockIdx.x * 256 + (wave & 3) * 64;
  const int n0     = blockIdx.y * 64 + (wave >> 2) * 32;

  v8f c[4][2];
#pragma unroll
  for (int nt = 0; nt < 2; ++nt) {
    const float bias = bo[n0 + nt * 16 + col];
#pragma unroll
    for (int i = 0; i < 4; ++i)
#pragma unroll
      for (int r = 0; r < 8; ++r) c[i][nt][r] = bias;
  }

  for (int k0 = 0; k0 < N_STATE; k0 += 32) {
    __builtin_prefetch(Wob + (size_t)(n0 + col) * N_STATE + k0 + 128, 0, 0);
    v16bf a[4];
#pragma unroll
    for (int i = 0; i < 4; ++i)
      a[i] = load_a_bf16(wv, N_STATE, m_base + 16 * i, k0, lane);
#pragma unroll
    for (int nt = 0; nt < 2; ++nt) {
      const v16bf b = load_bt_bf16(Wob, N_STATE, n0 + nt * 16, k0, lane);
#pragma unroll
      for (int i = 0; i < 4; ++i)
        c[i][nt] = __builtin_amdgcn_wmma_f32_16x16x32_bf16(
            false, a[i], false, b, (short)0, c[i][nt], false, false);
    }
  }

#pragma unroll
  for (int i = 0; i < 4; ++i)
#pragma unroll
    for (int nt = 0; nt < 2; ++nt)
#pragma unroll
      for (int r = 0; r < 8; ++r) {
        const int m = m_base + 16 * i + r + 8 * half;
        out[(size_t)m * N_STATE + n0 + nt * 16 + col] = c[i][nt][r];
      }
}

// ---------------------------------------------------------------------------
extern "C" void kernel_launch(void* const* d_in, const int* in_sizes, int n_in,
                              void* d_out, int out_size, void* d_ws, size_t ws_size,
                              hipStream_t stream) {
  (void)in_sizes; (void)n_in; (void)out_size; (void)ws_size;
  // setup_inputs order: x, mask, Wq, bq, Wk, Wv, bv, Wo, bo
  const float* x  = (const float*)d_in[0];
  // mask (d_in[1]) is applied analytically (causal -1e9)
  const float* Wq = (const float*)d_in[2];
  const float* bq = (const float*)d_in[3];
  const float* Wk = (const float*)d_in[4];
  const float* Wv = (const float*)d_in[5];
  const float* bv = (const float*)d_in[6];
  const float* Wo = (const float*)d_in[7];
  const float* bo = (const float*)d_in[8];
  float* out = (float*)d_out;

  // workspace layout (bf16 elems):
  //  [0,QEL)      Q [B,H,T,D]     [QEL,2Q)  K [B,H,T,D]
  //  [2Q,3Q)      V^T [B,H,D,T]   [3Q,4Q)   wv [BT,S]
  //  [4Q,5Q)      x_bf16          [5Q,+4W)  Wq,Wk,Wv,Wo bf16
  __bf16* wsb = (__bf16*)d_ws;
  __bf16* Qp  = wsb;
  __bf16* Kp  = wsb + QEL;
  __bf16* Vtp = wsb + 2 * QEL;
  __bf16* WVb = wsb + 3 * QEL;
  __bf16* xb  = wsb + 4 * QEL;
  __bf16* Wqb = wsb + 5 * QEL;
  __bf16* Wkb = Wqb + WEL;
  __bf16* Wvb = Wkb + WEL;
  __bf16* Wob = Wvb + WEL;

  // f32 -> bf16 conversions (packed cvt)
  {
    const int nx4 = (int)(QEL / 4);
    const int nw4 = (int)(WEL / 4);
    cvt_kernel<<<dim3((nx4 + 255) / 256), 256, 0, stream>>>(x,  xb,  nx4);
    cvt_kernel<<<dim3((nw4 + 255) / 256), 256, 0, stream>>>(Wq, Wqb, nw4);
    cvt_kernel<<<dim3((nw4 + 255) / 256), 256, 0, stream>>>(Wk, Wkb, nw4);
    cvt_kernel<<<dim3((nw4 + 255) / 256), 256, 0, stream>>>(Wv, Wvb, nw4);
    cvt_kernel<<<dim3((nw4 + 255) / 256), 256, 0, stream>>>(Wo, Wob, nw4);
  }

  qkv_kernel<<<dim3(BT / 256, 3 * N_STATE / 64), 256, 0, stream>>>(
      xb, Wqb, Wkb, Wvb, bq, bv, wsb /* qkv planes at offset 0 */);
  attn_kernel<<<dim3(BATCH * N_HEAD * (N_CTX / 16) / 8), 256, 0, stream>>>(
      Qp, Kp, Vtp, WVb);
  oproj_kernel<<<dim3(BT / 256, N_STATE / 64), 256, 0, stream>>>(
      WVb, Wob, bo, out);
}
